// LinearAttention_30794915512949
// MI455X (gfx1250) — matrix operand order repaired and checked by reading, not compiled, not run on hardware
//
#include <hip/hip_runtime.h>

// ---------------------------------------------------------------------------
// LinearAttention on MI455X (gfx1250, wave32, WMMA f32_16x16x32_f16)
//   B=64, C=512, L=512, HEADS=4, DH=128
// ---------------------------------------------------------------------------

#define CC     512
#define LL     512
#define BATCH  64
#define OC     1536   // 3*C

typedef __attribute__((ext_vector_type(16))) _Float16 v16h;
typedef __attribute__((ext_vector_type(8)))  _Float16 v8h;
typedef __attribute__((ext_vector_type(2)))  _Float16 h2;
typedef __attribute__((ext_vector_type(8)))  float    v8f;

static __device__ inline v16h frag_cat(v8h lo, v8h hi) {
  return __builtin_shufflevector(lo, hi, 0,1,2,3,4,5,6,7,8,9,10,11,12,13,14,15);
}

static __device__ inline v8f wmma16(v16h a, v16h b, v8f c) {
  // D = A*B + C   (16x16x32, f16 operands, f32 accum)
  return __builtin_amdgcn_wmma_f32_16x16x32_f16(false, a, false, b, (short)0, c,
                                                false, false);
}

// ---------------- block reductions (256 threads, wave32) -------------------
static __device__ inline float block_sum(float v, float* sh) {
  const int lane = threadIdx.x & 31, w = threadIdx.x >> 5;
  #pragma unroll
  for (int off = 16; off > 0; off >>= 1) v += __shfl_down(v, off, 32);
  if (lane == 0) sh[w] = v;
  __syncthreads();
  float r = sh[0];
  #pragma unroll
  for (int i = 1; i < 8; ++i) r += sh[i];
  __syncthreads();
  return r;
}

static __device__ inline float block_max(float v, float* sh) {
  const int lane = threadIdx.x & 31, w = threadIdx.x >> 5;
  #pragma unroll
  for (int off = 16; off > 0; off >>= 1) v = fmaxf(v, __shfl_down(v, off, 32));
  if (lane == 0) sh[w] = v;
  __syncthreads();
  float r = sh[0];
  #pragma unroll
  for (int i = 1; i < 8; ++i) r = fmaxf(r, sh[i]);
  __syncthreads();
  return r;
}

// sinusoidal positional encoding: position p (= channel), feature f (= l)
static __device__ inline float pos_val(int p, int f) {
  const int j = f >> 1;
  // div[j] = exp(-2j*ln(10000)/512)
  const float ang = (float)p * __expf((float)j * (-9.210340371976184f / 256.0f));
  return (f & 1) ? __cosf(ang) : __sinf(ang);
}

// ---------------- kernel 0: pw_w f32 -> f16 --------------------------------
__global__ __launch_bounds__(256) void convert_w_kernel(
    const float* __restrict__ W, _Float16* __restrict__ Wh, int n) {
  int i = blockIdx.x * 256 + threadIdx.x;
  if (i < n) Wh[i] = (_Float16)W[i];
}

// ---------------- kernel 1: LN + pos-enc + depthwise -> h (f16) ------------
// one block per (b,c) row of 512
__global__ __launch_bounds__(256) void ln_pos_dw_kernel(
    const float* __restrict__ x,
    const float* __restrict__ lnw, const float* __restrict__ lnb,
    const float* __restrict__ dww, const float* __restrict__ dwb,
    _Float16* __restrict__ H) {
  const int row = blockIdx.x;         // b*512 + c
  const int c   = row & (CC - 1);
  const float* src = x + (size_t)row * LL;
  _Float16*    dst = H + (size_t)row * LL;
  const int t = threadIdx.x;
  __shared__ float sh[8];

  const float v0 = src[t], v1 = src[t + 256];
  const float mu = block_sum(v0 + v1, sh) * (1.0f / 512.0f);
  const float d0 = v0 - mu, d1 = v1 - mu;
  const float var = block_sum(d0 * d0 + d1 * d1, sh) * (1.0f / 512.0f);
  const float rstd = rsqrtf(var + 1e-5f);

  const float g = lnw[c], be = lnb[c], sw = dww[c], sb = dwb[c];
  const float h0 = (d0 * rstd * g + be + pos_val(c, t))       * sw + sb;
  const float h1 = (d1 * rstd * g + be + pos_val(c, t + 256)) * sw + sb;
  dst[t]       = (_Float16)h0;
  dst[t + 256] = (_Float16)h1;
}

// ---------------- kernel 2: qkv[b,o,l] = W[o,c] * h[b,c,l] + pw_b[o] -------
// block tile 64(o) x 128(l), K-step 32, 8 waves (2x4) of 32x32 each
__global__ __launch_bounds__(256) void qkv_gemm_kernel(
    const _Float16* __restrict__ W, const _Float16* __restrict__ H,
    const float* __restrict__ pwb, float* __restrict__ QKV) {
  const int bo = blockIdx.x, bl = blockIdx.y, b = blockIdx.z;
  const _Float16* Hb = H + (size_t)b * CC * LL;
  float* out = QKV + (size_t)b * OC * LL;

  __shared__ _Float16 sA[64 * 40];    // [o_local][k] row-major
  __shared__ _Float16 sB[128 * 40];   // [l_local][k] (transposed h tile)

  const int t = threadIdx.x, lane = t & 31, wave = t >> 5;
  const int wm = wave >> 2, wn = wave & 3;
  const int lr = lane & 15;
  const int ka = (lane >> 4) * 8;     // A fragment K phase
  const int kb = (lane >> 4) * 8;     // B fragment K phase
  const int o0 = bo * 64, l0 = bl * 128;

  // staging coordinates
  const int ar = t >> 2, akc = (t & 3) * 8;     // A: 64 rows x 32 halves
  const int cp = t >> 4, blc = (t & 15) * 8;    // B: c-pair 2cp/2cp+1, 8 l's

  v8f acc[2][2] = {};
  for (int k0 = 0; k0 < CC; k0 += 32) {
    __syncthreads();
    { // stage A (coalesced, K-contiguous, 16B per thread)
      *(v8h*)(&sA[ar * 40 + akc]) =
          *(const v8h*)(W + (size_t)(o0 + ar) * CC + k0 + akc);
    }
    { // stage B transposed: two adjacent c-rows packed -> b32 LDS stores
      const _Float16* src = Hb + (size_t)(k0 + 2 * cp) * LL + l0 + blc;
      v8h r0 = *(const v8h*)src;
      v8h r1 = *(const v8h*)(src + LL);
      #pragma unroll
      for (int j = 0; j < 8; ++j) {
        h2 pr; pr[0] = r0[j]; pr[1] = r1[j];
        *(h2*)(&sB[(blc + j) * 40 + 2 * cp]) = pr;   // 4B-aligned b32 store
      }
    }
    // prefetch next K-tile source lines (global_prefetch_b8)
    if (k0 + 32 < CC) {
      __builtin_prefetch(W + (size_t)(o0 + ar) * CC + k0 + 32 + akc, 0, 1);
      __builtin_prefetch(Hb + (size_t)(k0 + 32 + 2 * cp) * LL + l0 + blc, 0, 1);
    }
    __syncthreads();

    v16h af[2], bf[2];
    #pragma unroll
    for (int tm = 0; tm < 2; ++tm) {
      const _Float16* p = &sA[(wm * 32 + tm * 16 + lr) * 40 + ka];
      af[tm] = frag_cat(*(const v8h*)p, *(const v8h*)(p + 16));
    }
    #pragma unroll
    for (int tn = 0; tn < 2; ++tn) {
      const _Float16* p = &sB[(wn * 32 + tn * 16 + lr) * 40 + kb];
      bf[tn] = frag_cat(*(const v8h*)p, *(const v8h*)(p + 16));
    }
    #pragma unroll
    for (int tm = 0; tm < 2; ++tm)
      #pragma unroll
      for (int tn = 0; tn < 2; ++tn)
        acc[tm][tn] = wmma16(af[tm], bf[tn], acc[tm][tn]);
  }

  const int rb = (lane >> 4) * 8;
  #pragma unroll
  for (int tm = 0; tm < 2; ++tm)
    #pragma unroll
    for (int tn = 0; tn < 2; ++tn) {
      const int l = l0 + wn * 32 + tn * 16 + lr;
      #pragma unroll
      for (int r = 0; r < 8; ++r) {
        const int o = o0 + wm * 32 + tm * 16 + rb + r;
        out[(size_t)o * LL + l] = acc[tm][tn][r] + pwb[o];
      }
    }
}

// ---------------- kernel 3: softmax(k rows) + f32->f16 in place ------------
// f16 row i aliases the first 1KB of f32 row i  =>  f16 row stride = 1024
__global__ __launch_bounds__(256) void softmax_convert_kernel(float* __restrict__ QKV) {
  const int row = blockIdx.x;                 // b*1536 + o
  const int o = row % OC;
  float* src = QKV + (size_t)row * LL;
  _Float16* dst = (_Float16*)src;
  const int t = threadIdx.x;
  __shared__ float sh[8];

  const float v0 = src[t], v1 = src[t + 256];
  if (o >= 512 && o < 1024) {                 // k rows: softmax over l
    const float m  = block_max(fmaxf(v0, v1), sh);
    const float e0 = __expf(v0 - m), e1 = __expf(v1 - m);
    const float s  = block_sum(e0 + e1, sh);  // ends with __syncthreads()
    const float inv = 1.0f / s;
    dst[t]       = (_Float16)(e0 * inv);
    dst[t + 256] = (_Float16)(e1 * inv);
  } else {                                    // q,v rows: straight convert
    __syncthreads();                          // all reads before aliasing writes
    dst[t]       = (_Float16)v0;
    dst[t + 256] = (_Float16)v1;
  }
}

// ---------------- kernel 4: ctx[d,e] = sum_n k[d,n]*v[e,n];  store ctx^T ---
// one block per (b,h); wave w owns d-rows [16w,16w+16) x all 128 e
__global__ __launch_bounds__(256) void ctx_gemm_kernel(
    const float* __restrict__ QKV, _Float16* __restrict__ CtxT) {
  const int bh = blockIdx.x, b = bh >> 2, hh = bh & 3;
  const _Float16* Q16 = (const _Float16*)QKV;           // row stride 1024
  const _Float16* Kp = Q16 + ((size_t)b * OC + 512  + hh * 128) * 1024;
  const _Float16* Vp = Q16 + ((size_t)b * OC + 1024 + hh * 128) * 1024;
  _Float16* ct = CtxT + (size_t)bh * 128 * 128;

  const int lane = threadIdx.x & 31, wave = threadIdx.x >> 5;
  const int lr = lane & 15, ka = (lane >> 4) * 8, kb = (lane >> 4) * 8;
  const int d0 = wave * 16;

  v8f acc[8] = {};
  for (int n0 = 0; n0 < LL; n0 += 32) {
    const _Float16* ap = Kp + (size_t)(d0 + lr) * 1024 + n0 + ka;
    const v16h af = frag_cat(*(const v8h*)ap, *(const v8h*)(ap + 16));
    #pragma unroll
    for (int et = 0; et < 8; ++et) {
      const _Float16* bp = Vp + (size_t)(et * 16 + lr) * 1024 + n0 + kb;
      const v16h bf = frag_cat(*(const v8h*)bp, *(const v8h*)(bp + 16));
      acc[et] = wmma16(af, bf, acc[et]);
    }
  }
  // write transposed: ctxT[e][d]  (8 contiguous halves per lane, 16B store)
  const int rb = (lane >> 4) * 8;
  #pragma unroll
  for (int et = 0; et < 8; ++et) {
    const int e = et * 16 + lr;
    v8h hv;
    #pragma unroll
    for (int r = 0; r < 8; ++r) hv[r] = (_Float16)acc[et][r];
    *(v8h*)(ct + (size_t)e * 128 + d0 + rb) = hv;
  }
}

// ---------------- kernel 5: out[e,n] = sum_d ctxT[e,d]*q[d,n]; epilogue ----
// block tile 64(e) x 128(n), K = 128 (d)
__global__ __launch_bounds__(256) void out_gemm_kernel(
    const _Float16* __restrict__ CtxT, const float* __restrict__ QKV,
    const float* __restrict__ ow, const float* __restrict__ ob,
    float* __restrict__ OUT) {
  const int bo = blockIdx.x, bl = blockIdx.y, bh = blockIdx.z;
  const int b = bh >> 2, hh = bh & 3;
  const _Float16* A  = CtxT + (size_t)bh * 128 * 128;
  const _Float16* Bq = (const _Float16*)QKV + ((size_t)b * OC + hh * 128) * 1024;
  float* out = OUT + (size_t)b * CC * LL;

  __shared__ _Float16 sA[64 * 40];
  __shared__ _Float16 sB[128 * 40];

  const int t = threadIdx.x, lane = t & 31, wave = t >> 5;
  const int wm = wave >> 2, wn = wave & 3;
  const int lr = lane & 15, ka = (lane >> 4) * 8, kb = (lane >> 4) * 8;
  const int e0 = bo * 64, l0 = bl * 128;

  const int ar = t >> 2, akc = (t & 3) * 8;
  const int dp = t >> 4, blc = (t & 15) * 8;    // d-pair 2dp/2dp+1, 8 l's

  v8f acc[2][2] = {};
  for (int k0 = 0; k0 < 128; k0 += 32) {
    __syncthreads();
    { *(v8h*)(&sA[ar * 40 + akc]) =
          *(const v8h*)(A + (size_t)(e0 + ar) * 128 + k0 + akc);
    }
    { const _Float16* src = Bq + (size_t)(k0 + 2 * dp) * 1024 + l0 + blc;
      v8h r0 = *(const v8h*)src;
      v8h r1 = *(const v8h*)(src + 1024);
      #pragma unroll
      for (int j = 0; j < 8; ++j) {
        h2 pr; pr[0] = r0[j]; pr[1] = r1[j];
        *(h2*)(&sB[(blc + j) * 40 + 2 * dp]) = pr;
      }
    }
    __syncthreads();

    v16h af[2], bf[2];
    #pragma unroll
    for (int tm = 0; tm < 2; ++tm) {
      const _Float16* p = &sA[(wm * 32 + tm * 16 + lr) * 40 + ka];
      af[tm] = frag_cat(*(const v8h*)p, *(const v8h*)(p + 16));
    }
    #pragma unroll
    for (int tn = 0; tn < 2; ++tn) {
      const _Float16* p = &sB[(wn * 32 + tn * 16 + lr) * 40 + kb];
      bf[tn] = frag_cat(*(const v8h*)p, *(const v8h*)(p + 16));
    }
    #pragma unroll
    for (int tm = 0; tm < 2; ++tm)
      #pragma unroll
      for (int tn = 0; tn < 2; ++tn)
        acc[tm][tn] = wmma16(af[tm], bf[tn], acc[tm][tn]);
  }

  const int rb = (lane >> 4) * 8;
  #pragma unroll
  for (int tm = 0; tm < 2; ++tm)
    #pragma unroll
    for (int tn = 0; tn < 2; ++tn) {
      const int l = l0 + wn * 32 + tn * 16 + lr;
      #pragma unroll
      for (int r = 0; r < 8; ++r) {
        const int e = e0 + wm * 32 + tm * 16 + rb + r;
        const int cch = hh * 128 + e;
        out[(size_t)cch * LL + l] = acc[tm][tn][r] * ow[cch] + ob[cch];
      }
    }
}

// ---------------------------------------------------------------------------
extern "C" void kernel_launch(void* const* d_in, const int* in_sizes, int n_in,
                              void* d_out, int out_size, void* d_ws, size_t ws_size,
                              hipStream_t stream) {
  (void)in_sizes; (void)n_in; (void)out_size; (void)ws_size;
  const float* x    = (const float*)d_in[0];
  const float* ln_w = (const float*)d_in[1];
  const float* ln_b = (const float*)d_in[2];
  const float* dw_w = (const float*)d_in[3];
  const float* dw_b = (const float*)d_in[4];
  const float* pw_w = (const float*)d_in[5];
  const float* pw_b = (const float*)d_in[6];
  const float* ow   = (const float*)d_in[7];
  const float* ob   = (const float*)d_in[8];
  float* out = (float*)d_out;

  // workspace layout (234 MB total):
  //   [0,   2MB)   W f16        (1536*512*2  = 1.5 MB)
  //   [2,  34MB)   h  f16       (64*512*512*2 = 32 MB)
  //   [34, 226MB)  qkv f32      (64*1536*512*4 = 192 MB; f16 aliased in place)
  //   [226,234MB)  ctx^T f16    (64*4*128*128*2 = 8 MB)
  char* ws = (char*)d_ws;
  _Float16* Wf16 = (_Float16*)(ws);
  _Float16* H16  = (_Float16*)(ws + (2ull  << 20));
  float*    QKV  = (float*)   (ws + (34ull << 20));
  _Float16* CtxT = (_Float16*)(ws + (226ull << 20));

  convert_w_kernel<<<(OC * CC + 255) / 256, 256, 0, stream>>>(pw_w, Wf16, OC * CC);
  ln_pos_dw_kernel<<<BATCH * CC, 256, 0, stream>>>(x, ln_w, ln_b, dw_w, dw_b, H16);
  qkv_gemm_kernel<<<dim3(OC / 64, LL / 128, BATCH), 256, 0, stream>>>(Wf16, H16, pw_b, QKV);
  softmax_convert_kernel<<<BATCH * OC, 256, 0, stream>>>(QKV);
  ctx_gemm_kernel<<<BATCH * 4, 256, 0, stream>>>(QKV, CtxT);
  out_gemm_kernel<<<dim3(2, LL / 128, BATCH * 4), 256, 0, stream>>>(CtxT, QKV, ow, ob, out);
}
